// CausalAttention_84963043049620
// MI455X (gfx1250) — matrix operand order repaired and checked by reading, not compiled, not run on hardware
//
#include <hip/hip_runtime.h>
#include <hip/hip_bf16.h>

// ---------------------------------------------------------------------------
// Causal attention forward, MI455X (gfx1250), wave32.
// bf16 WMMA (v_wmma_f32_16x16x32_bf16) + async global->LDS staging (ASYNCcnt).
// ---------------------------------------------------------------------------

typedef __bf16 bf16;
typedef __attribute__((ext_vector_type(8)))  bf16  v8bf;
typedef __attribute__((ext_vector_type(16))) bf16  v16bf;
typedef __attribute__((ext_vector_type(8)))  float v8f;

#define NB_B   4
#define NB_S   2048
#define NB_D   768
#define NB_H   12
#define NB_HD  64
#define NB_TD  (3 * NB_D)          // 2304
#define NB_ROWS (NB_B * NB_S)      // 8192

// -------- async global->LDS (gfx1250 GLOBAL_LOAD_ASYNC_TO_LDS_B128) --------
#if defined(__has_builtin)
#  if __has_builtin(__builtin_amdgcn_global_load_async_to_lds_b128)
#    define HAS_ASYNC_LDS 1
#  endif
#endif
#ifndef HAS_ASYNC_LDS
#  define HAS_ASYNC_LDS 0
#endif

typedef int v4i_vec __attribute__((vector_size(16)));
typedef __attribute__((address_space(1))) v4i_vec* gb128_t;   // global int4*
typedef __attribute__((address_space(3))) v4i_vec* sb128_t;   // LDS int4*

__device__ __forceinline__ void cp16_g2s(const bf16* g, bf16* l) {
#if HAS_ASYNC_LDS
    __builtin_amdgcn_global_load_async_to_lds_b128((gb128_t)g, (sb128_t)l, 0, 0);
#else
    *(v8bf*)l = *(const v8bf*)g;     // sync fallback: global load + ds store
#endif
}

__device__ __forceinline__ void async_wait() {
#if HAS_ASYNC_LDS
    asm volatile("s_wait_asynccnt 0" ::: "memory");
#endif
}

// ---------------------------------------------------------------------------
__device__ __forceinline__ v16bf bf_join(v8bf lo, v8bf hi) {
    return __builtin_shufflevector(lo, hi, 0, 1, 2, 3, 4, 5, 6, 7,
                                           8, 9, 10, 11, 12, 13, 14, 15);
}

// A fragment (16x32, row-major A[M x K]).  lane={m=l&15, kh=l>>4};
// K = e + 16*kh (contiguous, matches frag16).  Two contiguous 16B loads.
__device__ __forceinline__ v16bf load_a_frag(const bf16* __restrict__ base,
                                             int ld, int row0, int k0, int lane) {
    int m = lane & 15, kh = lane >> 4;
    const bf16* p = base + (size_t)(row0 + m) * ld + k0 + 16 * kh;
    v8bf lo = *(const v8bf*)p;
    v8bf hi = *(const v8bf*)(p + 8);
    return bf_join(lo, hi);
}

// B fragment: 16 contiguous bf16 (K = e + 16*kh) from any pointer (LDS or global).
__device__ __forceinline__ v16bf frag16(const bf16* p) {
    v8bf lo = *(const v8bf*)p;
    v8bf hi = *(const v8bf*)(p + 8);
    return bf_join(lo, hi);
}

__device__ __forceinline__ v8f wmma_bf16(v16bf a, v16bf b, v8f c) {
    return __builtin_amdgcn_wmma_f32_16x16x32_bf16(
        false, a, false, b, (short)0, c, false, false);
}

// ---------------------------------------------------------------------------
// Prep kernels
// ---------------------------------------------------------------------------
__global__ void cvt_f32_bf16_kernel(const float* __restrict__ in,
                                    bf16* __restrict__ out, int n) {
    int i = (blockIdx.x * blockDim.x + threadIdx.x) * 4;
    if (i + 3 < n) {
        float4 f = *(const float4*)(in + i);
        out[i + 0] = (bf16)f.x;
        out[i + 1] = (bf16)f.y;
        out[i + 2] = (bf16)f.z;
        out[i + 3] = (bf16)f.w;
    }
}

// wt[n*K + k] = w[k*N + n]  (f32 -> bf16)
__global__ void transpose_cvt_kernel(const float* __restrict__ w,
                                     bf16* __restrict__ wt, int K, int N) {
    int idx = blockIdx.x * blockDim.x + threadIdx.x;
    if (idx >= K * N) return;
    int nrow = idx / K;
    int k    = idx % K;
    wt[idx] = (bf16)w[(size_t)k * N + nrow];
}

// ---------------------------------------------------------------------------
// QKV GEMM: 1 block = 128(M) x 64(N); Bt tile staged (async, double-buffered)
// in LDS once per block; A fragments direct from global.
// Epilogue scatters q,k -> [B,H,S,64]; v -> [B,H,64,S] (transposed).
// ---------------------------------------------------------------------------
__global__ void __launch_bounds__(256)
gemm_qkv_kernel(const bf16* __restrict__ A, const bf16* __restrict__ Bt,
                const float* __restrict__ bias,
                bf16* __restrict__ qb, bf16* __restrict__ kb,
                bf16* __restrict__ vtb) {
    __shared__ __align__(16) bf16 bsh[2][64][32];      // double-buffered B tile

    const int NBLK = NB_TD / 64;                       // 36
    int mb = blockIdx.x / NBLK, nb = blockIdx.x % NBLK;
    int lane = threadIdx.x & 31, w = threadIdx.x >> 5;
    int r0 = mb * 128 + w * 16;
    int c0 = nb * 64;
    int n = lane & 15, kh = lane >> 4;
    int tr = threadIdx.x >> 2, tc = (threadIdx.x & 3) * 8;   // staging coords

    cp16_g2s(Bt + (size_t)(c0 + tr) * NB_D + tc, &bsh[0][tr][tc]);
    async_wait();
    __syncthreads();

    v8f acc[4] = {};
    int buf = 0;
#pragma unroll 2
    for (int k0 = 0; k0 < NB_D; k0 += 32) {
        if (k0 + 32 < NB_D)   // issue next tile before computing on current
            cp16_g2s(Bt + (size_t)(c0 + tr) * NB_D + (k0 + 32) + tc,
                     &bsh[buf ^ 1][tr][tc]);
        v16bf a = load_a_frag(A, NB_D, r0, k0, lane);
#pragma unroll
        for (int t = 0; t < 4; ++t) {
            v16bf b = frag16(&bsh[buf][16 * t + n][16 * kh]);
            acc[t] = wmma_bf16(a, b, acc[t]);
        }
        async_wait();
        __syncthreads();
        buf ^= 1;
    }

#pragma unroll
    for (int t = 0; t < 4; ++t) {
        int col   = c0 + 16 * t + n;                   // [0,2304)
        int which = col / NB_D;                        // 0=q 1=k 2=v (uniform)
        int d     = col % NB_D;
        int h     = d >> 6, di = d & 63;
        float bv  = bias[col];
#pragma unroll
        for (int e = 0; e < 8; ++e) {
            int row = r0 + e + 8 * kh;                 // [0,8192)
            int bb  = row >> 11, s = row & (NB_S - 1);
            bf16 val = (bf16)(acc[t][e] + bv);
            size_t bh = (size_t)(bb * NB_H + h);
            if (which == 0)
                qb[(bh * NB_S + s) * NB_HD + di] = val;
            else if (which == 1)
                kb[(bh * NB_S + s) * NB_HD + di] = val;
            else
                vtb[(bh * NB_HD + di) * NB_S + s] = val;
        }
    }
}

// ---------------------------------------------------------------------------
// Flash-style causal attention. 1 block = 128 consecutive queries of one (b,h),
// 8 waves x 16 queries. K/V key-tiles of 32 staged once per block in LDS
// (async, double-buffered); all waves compute from LDS.
// ---------------------------------------------------------------------------
__global__ void __launch_bounds__(256)
attn_kernel(const bf16* __restrict__ qb, const bf16* __restrict__ kb,
            const bf16* __restrict__ vtb, bf16* __restrict__ yb) {
    __shared__ __align__(16) bf16 ksh[2][32][64];      // keys x hd
    __shared__ __align__(16) bf16 vsh[2][64][32];      // hd x keys (V^T)
    __shared__ __align__(16) bf16 plds[8][16][32];     // per-wave P C->A relayout

    int tid = threadIdx.x;
    int lane = tid & 31, wl = tid >> 5;
    int bh   = blockIdx.x >> 4;                        // 0..47
    int qblk = blockIdx.x & 15;
    int bb = bh / NB_H, h = bh % NB_H;
    int q0_blk = qblk * 128;
    int q0 = q0_blk + wl * 16;
    int myLast = q0 + 15;
    int lastBlk = q0_blk + 127;

    const bf16* qp = qb  + (size_t)bh * NB_S * NB_HD;
    const bf16* kp = kb  + (size_t)bh * NB_S * NB_HD;
    const bf16* vp = vtb + (size_t)bh * NB_HD * NB_S;
    int n = lane & 15, kh = lane >> 4;

    int krow = tid >> 3, kcol = (tid & 7) * 8;         // K tile: 32 x 64
    int vrow = tid >> 2, vcol = (tid & 3) * 8;         // V^T tile: 64 x 32

    v16bf qf0 = load_a_frag(qp, NB_HD, q0, 0, lane);
    v16bf qf1 = load_a_frag(qp, NB_HD, q0, 32, lane);

    v8f o[4] = {};
    float rmax[8], rsum[8];
#pragma unroll
    for (int e = 0; e < 8; ++e) { rmax[e] = -__builtin_inff(); rsum[e] = 0.f; }

    // stage first key tile
    cp16_g2s(kp + (size_t)(0 + krow) * NB_HD + kcol, &ksh[0][krow][kcol]);
    cp16_g2s(vp + (size_t)vrow * NB_S + 0 + vcol,    &vsh[0][vrow][vcol]);
    async_wait();
    __syncthreads();

    const float scale = 0.125f;                        // 1/sqrt(64)
    int buf = 0;
#pragma unroll 2
    for (int kk = 0; kk <= lastBlk; kk += 32) {
        if (kk + 32 <= lastBlk) {                      // prefetch next tile
            cp16_g2s(kp + (size_t)(kk + 32 + krow) * NB_HD + kcol,
                     &ksh[buf ^ 1][krow][kcol]);
            cp16_g2s(vp + (size_t)vrow * NB_S + (kk + 32) + vcol,
                     &vsh[buf ^ 1][vrow][vcol]);
        }

        if (kk <= myLast) {                            // causal: skip if all masked
            // ---- scores S[16q x 32k] = Q @ K^T (B-fragments from LDS)
            v8f s0 = {}, s1 = {};
            s0 = wmma_bf16(qf0, frag16(&ksh[buf][n][0  + 16 * kh]), s0);
            s0 = wmma_bf16(qf1, frag16(&ksh[buf][n][32 + 16 * kh]), s0);
            s1 = wmma_bf16(qf0, frag16(&ksh[buf][16 + n][0  + 16 * kh]), s1);
            s1 = wmma_bf16(qf1, frag16(&ksh[buf][16 + n][32 + 16 * kh]), s1);

            // ---- scale + causal mask + online softmax (rows span 16-lane halves)
#pragma unroll
            for (int e = 0; e < 8; ++e) {
                int qrow = q0 + e + 8 * kh;
                float a0 = s0[e] * scale;
                float a1 = s1[e] * scale;
                if (kk + n > qrow)      a0 = -__builtin_inff();
                if (kk + 16 + n > qrow) a1 = -__builtin_inff();

                float mx = fmaxf(a0, a1);
                mx = fmaxf(mx, __shfl_xor(mx, 1));
                mx = fmaxf(mx, __shfl_xor(mx, 2));
                mx = fmaxf(mx, __shfl_xor(mx, 4));
                mx = fmaxf(mx, __shfl_xor(mx, 8));
                float mnew = fmaxf(rmax[e], mx);
                float fac  = __expf(rmax[e] - mnew);
                rmax[e] = mnew;

                float p0 = __expf(a0 - mnew);
                float p1 = __expf(a1 - mnew);
                s0[e] = p0; s1[e] = p1;

                float ps = p0 + p1;
                ps += __shfl_xor(ps, 1);
                ps += __shfl_xor(ps, 2);
                ps += __shfl_xor(ps, 4);
                ps += __shfl_xor(ps, 8);
                rsum[e] = rsum[e] * fac + ps;

#pragma unroll
                for (int t = 0; t < 4; ++t) o[t][e] *= fac;
            }

            // ---- relayout P (C layout -> A layout) via per-wave LDS
#pragma unroll
            for (int e = 0; e < 8; ++e) {
                int row = e + 8 * kh;
                plds[wl][row][n]      = (bf16)s0[e];
                plds[wl][row][n + 16] = (bf16)s1[e];
            }
            // DS ops are in-order per wave; counter wait suffices (no barrier).
            asm volatile("s_wait_dscnt 0" ::: "memory");
            v16bf pf = bf_join(*(const v8bf*)&plds[wl][n][16 * kh],
                               *(const v8bf*)&plds[wl][n][16 * kh + 8]);

            // ---- O += P @ V (V^T fragments from LDS)
#pragma unroll
            for (int t = 0; t < 4; ++t) {
                v16bf vf = frag16(&vsh[buf][16 * t + n][16 * kh]);
                o[t] = wmma_bf16(pf, vf, o[t]);
            }
        }

        async_wait();          // next tile landed (this wave's async ops)
        __syncthreads();       // everyone done reading current tile
        buf ^= 1;
    }

    // ---- normalize and write y as [B,S,D] bf16
#pragma unroll
    for (int t = 0; t < 4; ++t) {
#pragma unroll
        for (int e = 0; e < 8; ++e) {
            int s = q0 + e + 8 * kh;
            float val = o[t][e] / rsum[e];
            yb[((size_t)bb * NB_S + s) * NB_D + h * NB_HD + 16 * t + n] = (bf16)val;
        }
    }
}

// ---------------------------------------------------------------------------
// Output projection: out[8192 x 768] = y @ W_proj + b_proj (f32 out).
// Same LDS-staged structure as the QKV GEMM.
// ---------------------------------------------------------------------------
__global__ void __launch_bounds__(256)
gemm_proj_kernel(const bf16* __restrict__ A, const bf16* __restrict__ Bt,
                 const float* __restrict__ bias, float* __restrict__ out) {
    __shared__ __align__(16) bf16 bsh[2][64][32];

    const int NBLK = NB_D / 64;                        // 12
    int mb = blockIdx.x / NBLK, nb = blockIdx.x % NBLK;
    int lane = threadIdx.x & 31, w = threadIdx.x >> 5;
    int r0 = mb * 128 + w * 16;
    int c0 = nb * 64;
    int n = lane & 15, kh = lane >> 4;
    int tr = threadIdx.x >> 2, tc = (threadIdx.x & 3) * 8;

    cp16_g2s(Bt + (size_t)(c0 + tr) * NB_D + tc, &bsh[0][tr][tc]);
    async_wait();
    __syncthreads();

    v8f acc[4] = {};
    int buf = 0;
#pragma unroll 2
    for (int k0 = 0; k0 < NB_D; k0 += 32) {
        if (k0 + 32 < NB_D)
            cp16_g2s(Bt + (size_t)(c0 + tr) * NB_D + (k0 + 32) + tc,
                     &bsh[buf ^ 1][tr][tc]);
        v16bf a = load_a_frag(A, NB_D, r0, k0, lane);
#pragma unroll
        for (int t = 0; t < 4; ++t) {
            v16bf b = frag16(&bsh[buf][16 * t + n][16 * kh]);
            acc[t] = wmma_bf16(a, b, acc[t]);
        }
        async_wait();
        __syncthreads();
        buf ^= 1;
    }

#pragma unroll
    for (int t = 0; t < 4; ++t) {
        int col  = c0 + 16 * t + n;
        float bv = bias[col];
#pragma unroll
        for (int e = 0; e < 8; ++e) {
            int row = r0 + e + 8 * kh;
            out[(size_t)row * NB_D + col] = acc[t][e] + bv;
        }
    }
}

// ---------------------------------------------------------------------------
// Host-side launch
// ---------------------------------------------------------------------------
extern "C" void kernel_launch(void* const* d_in, const int* in_sizes, int n_in,
                              void* d_out, int out_size, void* d_ws, size_t ws_size,
                              hipStream_t stream) {
    const float* x      = (const float*)d_in[0];
    const float* W_attn = (const float*)d_in[1];
    const float* b_attn = (const float*)d_in[2];
    const float* W_proj = (const float*)d_in[3];
    const float* b_proj = (const float*)d_in[4];
    float* out = (float*)d_out;

    char* ws = (char*)d_ws;
    size_t off = 0;
    auto carve = [&](size_t bytes) -> char* {
        char* p = ws + off;
        off = (off + bytes + 255) & ~(size_t)255;
        return p;
    };

    const size_t SZ_XB  = (size_t)NB_ROWS * NB_D * sizeof(bf16);
    const size_t SZ_WA  = (size_t)NB_TD * NB_D * sizeof(bf16);
    const size_t SZ_WP  = (size_t)NB_D * NB_D * sizeof(bf16);
    const size_t SZ_QKV = (size_t)NB_B * NB_H * NB_S * NB_HD * sizeof(bf16);

    bf16* xb      = (bf16*)carve(SZ_XB);
    bf16* wattn_t = (bf16*)carve(SZ_WA);
    bf16* wproj_t = (bf16*)carve(SZ_WP);
    bf16* qbuf    = (bf16*)carve(SZ_QKV);
    bf16* kbuf    = (bf16*)carve(SZ_QKV);
    bf16* vtbuf   = (bf16*)carve(SZ_QKV);
    bf16* ybuf    = xb;   // x-bf16 dead after QKV GEMM; reuse for y
    (void)ws_size; (void)in_sizes; (void)n_in; (void)out_size;

    {   // 1) x -> bf16
        int n = NB_ROWS * NB_D;
        cvt_f32_bf16_kernel<<<n / 4 / 256, 256, 0, stream>>>(x, xb, n);
    }
    {   // 2) transpose+convert weights
        int n = NB_D * NB_TD;
        transpose_cvt_kernel<<<(n + 255) / 256, 256, 0, stream>>>(W_attn, wattn_t, NB_D, NB_TD);
    }
    {
        int n = NB_D * NB_D;
        transpose_cvt_kernel<<<(n + 255) / 256, 256, 0, stream>>>(W_proj, wproj_t, NB_D, NB_D);
    }
    // 3) QKV GEMM (64 x 36 blocks)
    gemm_qkv_kernel<<<(NB_ROWS / 128) * (NB_TD / 64), 256, 0, stream>>>(
        xb, wattn_t, b_attn, qbuf, kbuf, vtbuf);
    // 4) causal attention (48 bh x 16 q-blocks)
    attn_kernel<<<NB_B * NB_H * (NB_S / 128), 256, 0, stream>>>(
        qbuf, kbuf, vtbuf, ybuf);
    // 5) output projection (64 x 12 blocks)
    gemm_proj_kernel<<<(NB_ROWS / 128) * (NB_D / 64), 256, 0, stream>>>(
        ybuf, wproj_t, b_proj, out);
}
